// RoPEMultiHeadSelfAttention_85246510891660
// MI455X (gfx1250) — matrix operand order repaired and checked by reading, not compiled, not run on hardware
//
#include <hip/hip_runtime.h>
#include <hip/hip_bf16.h>

typedef __attribute__((ext_vector_type(16))) _Float16 v16h;
typedef __attribute__((ext_vector_type(8)))  _Float16 v8h;
typedef __attribute__((ext_vector_type(8)))  float    v8f;

#define WMMA_F16(a, b, c) \
    __builtin_amdgcn_wmma_f32_16x16x32_f16(false, (a), false, (b), (short)0, (c), false, false)

// ---- compile-time problem dims -------------------------------------------
constexpr int CB = 4, CS = 2048, CD = 1024, CH = 16, CHd = 64;
constexpr int CM = CB * CS;   // 8192 rows

// ---- helpers --------------------------------------------------------------
__device__ inline v8h cvt8h(const float* s) {
    float4 a = *(const float4*)s;
    float4 b = *(const float4*)(s + 4);
    v8h r;
    r[0] = (_Float16)a.x; r[1] = (_Float16)a.y;
    r[2] = (_Float16)a.z; r[3] = (_Float16)a.w;
    r[4] = (_Float16)b.x; r[5] = (_Float16)b.y;
    r[6] = (_Float16)b.z; r[7] = (_Float16)b.w;
    return r;
}

// A-matrix 16x32 frag: lane row=lane%16, kb=(lane<16)?0:16,
// elems j -> K=kb+j (contiguous). Works on LDS or global f16.
__device__ inline v16h load_a_frag(const _Float16* base, int stride, int row,
                                   int koff, int kb) {
    const _Float16* r = base + (size_t)row * stride + koff + kb;
    v16h f;
#pragma unroll
    for (int j = 0; j < 8; j += 2) {
        union { unsigned int u; _Float16 h[2]; } t0, t1;
        t0.u = *(const unsigned int*)(r + j);
        t1.u = *(const unsigned int*)(r + 8 + j);
        f[j]     = t0.h[0]; f[j + 1] = t0.h[1];
        f[8 + j] = t1.h[0]; f[9 + j] = t1.h[1];
    }
    return f;
}

// B-matrix 32x16 frag: lane col=lane%16, elems j -> K=16*(lane/16)+j
// (caller folds 16*(lane/16) into koff; 16B-aligned contiguous load)
__device__ inline v16h load_b_frag(const _Float16* base, int stride, int col,
                                   int koff) {
    const v8h* p = (const v8h*)(base + col * stride + koff);
    v8h lo = p[0], hi = p[1];
    return __builtin_shufflevector(lo, hi, 0, 1, 2, 3, 4, 5, 6, 7,
                                   8, 9, 10, 11, 12, 13, 14, 15);
}

// async global->LDS copy of 16 bytes (CDNA5, tracked by ASYNCcnt)
__device__ inline void async_copy_b128(const _Float16* g, void* lds) {
    unsigned loff = (unsigned)(size_t)lds;   // low 32 bits = LDS byte offset
    asm volatile("global_load_async_to_lds_b128 %0, %1, off"
                 :: "v"(loff), "v"(g) : "memory");
}
template <int NCNT>
__device__ inline void async_wait_le() {
    asm volatile("s_wait_asynccnt %0" :: "n"(NCNT) : "memory");
}

// ---- all-f16 GEMM with async double-buffered staging ----------------------
// C[M,1024] = A[M,1024] @ BT[1024,1024]^T + bias.  256 thr = 8 waves (4x2),
// tile 128x128, BK=32, each wave 32x64 (8 WMMA / K-step).
// MODE 0: f16 store [M,N]   MODE 1: f16 store V-transposed [B,H,64,S]
// MODE 2: f32 store [M,N] * key_padding_mask[row]
template <int MODE>
__global__ __launch_bounds__(256) void gemm_f16_async(
    const _Float16* __restrict__ A, const _Float16* __restrict__ BT,
    const float* __restrict__ bias, void* __restrict__ Cout,
    const unsigned char* __restrict__ mask) {
    constexpr int K = CD, N = CD;
    __shared__ alignas(16) _Float16 As[2][128][40];   // [m][k]
    __shared__ alignas(16) _Float16 Bs[2][128][40];   // [n][k]

    const int tid = threadIdx.x;
    const int wave = tid >> 5, lane = tid & 31;
    const int lrow = lane & 15, halfid = lane >> 4;
    const int kb = halfid ? 16 : 0;
    const int wm = wave & 3, wn = wave >> 2;
    const int bm = blockIdx.x * 128, bn = blockIdx.y * 128;

    auto issue = [&](int buf, int k0) {
#pragma unroll
        for (int p = 0; p < 2; ++p) {
            int seg = tid + p * 256;                 // 0..511
            int r = seg >> 2, c8 = (seg & 3) * 8;
            async_copy_b128(A + (size_t)(bm + r) * K + k0 + c8,
                            &As[buf][r][c8]);
            async_copy_b128(BT + (size_t)(bn + r) * K + k0 + c8,
                            &Bs[buf][r][c8]);
        }
    };

    v8f acc[2][4];
#pragma unroll
    for (int mi = 0; mi < 2; ++mi)
#pragma unroll
        for (int ng = 0; ng < 4; ++ng)
#pragma unroll
            for (int v = 0; v < 8; ++v) acc[mi][ng][v] = 0.f;

    auto compute = [&](int buf) {
        v16h af0 = load_a_frag(&As[buf][0][0], 40, wm * 32 + lrow, 0, kb);
        v16h af1 = load_a_frag(&As[buf][0][0], 40, wm * 32 + 16 + lrow, 0, kb);
#pragma unroll
        for (int ng = 0; ng < 4; ++ng) {
            v16h bf = load_b_frag(&Bs[buf][0][0], 40,
                                  wn * 64 + ng * 16 + lrow, 16 * halfid);
            acc[0][ng] = WMMA_F16(af0, bf, acc[0][ng]);
            acc[1][ng] = WMMA_F16(af1, bf, acc[1][ng]);
        }
    };

    issue(0, 0);
#pragma unroll 1
    for (int k0 = 0; k0 < K - 32; k0 += 32) {      // steady state: prefetch
        int buf = (k0 >> 5) & 1;
        issue(buf ^ 1, k0 + 32);
        async_wait_le<4>();        // current buffer's 4 copies complete
        __syncthreads();
        compute(buf);
        __syncthreads();           // done reading before next overwrite
    }
    async_wait_le<0>();            // last tile
    __syncthreads();
    compute(((K >> 5) - 1) & 1);

#pragma unroll
    for (int mi = 0; mi < 2; ++mi)
#pragma unroll
        for (int ng = 0; ng < 4; ++ng) {
            int col = bn + wn * 64 + ng * 16 + lrow;
            float bv = bias[col];
#pragma unroll
            for (int v = 0; v < 8; ++v) {
                int row = bm + wm * 32 + mi * 16 + v + 8 * halfid;
                float val = acc[mi][ng][v] + bv;
                if constexpr (MODE == 0) {
                    ((_Float16*)Cout)[(size_t)row * N + col] = (_Float16)val;
                } else if constexpr (MODE == 1) {
                    int b = row >> 11, s = row & (CS - 1);
                    int h = col >> 6, d = col & 63;
                    ((_Float16*)Cout)[(((size_t)b * CH + h) * CHd + d) * CS + s]
                        = (_Float16)val;
                } else {
                    float mv = mask[row] ? 1.f : 0.f;
                    ((float*)Cout)[(size_t)row * N + col] = val * mv;
                }
            }
        }
}

// ---- fp32 -> f16 bulk convert (vectorized) --------------------------------
__global__ __launch_bounds__(256) void cvt_x_kernel(
    const float* __restrict__ src, _Float16* __restrict__ dst, int n8) {
    int idx = blockIdx.x * blockDim.x + threadIdx.x;   // over n/8
    if (idx >= n8) return;
    *(v8h*)(dst + (size_t)idx * 8) = cvt8h(src + (size_t)idx * 8);
}

// ---- weight fp32 [K,N] -> f16 transposed [N,K] (N=K=1024) -----------------
__global__ __launch_bounds__(256) void cvt_wT_kernel(
    const float* __restrict__ w, _Float16* __restrict__ wT) {
    int idx = blockIdx.x * blockDim.x + threadIdx.x;   // over 1024*1024
    if (idx >= CD * CD) return;
    int k = idx & (CD - 1), nn = idx >> 10;
    wT[idx] = (_Float16)w[(size_t)k * CD + nn];
}

// ---- RoPE in place on f16 Q,K ([B*S, D]) ----------------------------------
__global__ __launch_bounds__(256) void rope_inplace_kernel(
    _Float16* __restrict__ Qh, _Float16* __restrict__ Kh) {
    int idx = blockIdx.x * blockDim.x + threadIdx.x;   // over M * D/2
    if (idx >= CM * (CD / 2)) return;
    int row = idx >> 9;              // D/2 = 512
    int r   = idx & 511;
    int h = r >> 5, d = r & 31;
    int s = row & (CS - 1);
    float inv = __expf(-(float)d * (9.2103403719761836f / 32.f)); // 1e4^(-d/32)
    float fr = (float)s * inv;
    float sn, cs;
    __sincosf(fr, &sn, &cs);
    size_t base = (size_t)row * CD + h * 64 + d;
    float q1 = (float)Qh[base], q2 = (float)Qh[base + 32];
    Qh[base]      = (_Float16)(q1 * cs - q2 * sn);
    Qh[base + 32] = (_Float16)(q2 * cs + q1 * sn);
    float k1 = (float)Kh[base], k2 = (float)Kh[base + 32];
    Kh[base]      = (_Float16)(k1 * cs - k2 * sn);
    Kh[base + 32] = (_Float16)(k2 * cs + k1 * sn);
}

// ---- Flash attention: block = (qb, h, b), 4 waves x 16 q-rows -------------
// Double-buffered async K/V tile staging (s_wait_asynccnt 8 steady state).
__global__ __launch_bounds__(128) void attn_wmma(
    const _Float16* __restrict__ Qh, const _Float16* __restrict__ Kh,
    const _Float16* __restrict__ Vt, const unsigned char* __restrict__ mask,
    _Float16* __restrict__ ctx) {
    constexpr int S = CS, D = CD, Hd = CHd;
    __shared__ alignas(16) _Float16 Ks[2][64][72];     // [key][d]
    __shared__ alignas(16) _Float16 Vs[2][64][72];     // [d][key]
    __shared__ alignas(16) _Float16 Ps[4][16][72];     // per-wave P [row][key]
    __shared__ float maskAdd[2][64];

    const int tid = threadIdx.x;
    const int wave = tid >> 5, lane = tid & 31;
    const int lrow = lane & 15, halfid = lane >> 4;
    const int kb = halfid ? 16 : 0;
    const int qb = blockIdx.x, h = blockIdx.y, b = blockIdx.z;
    const int q0 = qb * 64 + wave * 16;

    const _Float16* qbase = Qh + (size_t)(b * S + q0) * D + h * Hd;
    v16h qf[2];
#pragma unroll
    for (int c = 0; c < 2; ++c)
        qf[c] = load_a_frag(qbase, D, lrow, c * 32, kb);

    float m_run[8], l_run[8];
    v8f o[4];
#pragma unroll
    for (int v = 0; v < 8; ++v) { m_run[v] = -1e30f; l_run[v] = 0.f; }
#pragma unroll
    for (int ng = 0; ng < 4; ++ng)
#pragma unroll
        for (int v = 0; v < 8; ++v) o[ng][v] = 0.f;

    const size_t vtHead = ((size_t)(b * CH + h)) * Hd;

    auto stage = [&](int buf, int s0) {
#pragma unroll
        for (int p = 0; p < 4; ++p) {
            int seg = tid + p * 128;                    // 0..511
            int r = seg >> 3, c8 = (seg & 7) * 8;
            async_copy_b128(Kh + (size_t)(b * S + s0 + r) * D + h * Hd + c8,
                            &Ks[buf][r][c8]);
            async_copy_b128(Vt + (vtHead + r) * S + s0 + c8,
                            &Vs[buf][r][c8]);
        }
        if (tid < 64)
            maskAdd[buf][tid] = mask[b * S + s0 + tid] ? 0.f : -1e30f;
    };

    auto compute_tile = [&](int buf) {
        // scores
        v8f sf[4];
#pragma unroll
        for (int kg = 0; kg < 4; ++kg) {
#pragma unroll
            for (int v = 0; v < 8; ++v) sf[kg][v] = 0.f;
#pragma unroll
            for (int c = 0; c < 2; ++c) {
                v16h bf = load_b_frag(&Ks[buf][0][0], 72, kg * 16 + lrow,
                                      c * 32 + 16 * halfid);
                sf[kg] = WMMA_F16(qf[c], bf, sf[kg]);
            }
            float madd = maskAdd[buf][kg * 16 + lrow];
#pragma unroll
            for (int v = 0; v < 8; ++v)
                sf[kg][v] = sf[kg][v] * 0.125f + madd;
        }

        // online softmax (row = v + 8*halfid; reduce across 16-lane half)
#pragma unroll
        for (int v = 0; v < 8; ++v) {
            float mx = sf[0][v];
#pragma unroll
            for (int kg = 1; kg < 4; ++kg) mx = fmaxf(mx, sf[kg][v]);
#pragma unroll
            for (int off = 1; off < 16; off <<= 1)
                mx = fmaxf(mx, __shfl_xor(mx, off, 16));
            float mnew  = fmaxf(m_run[v], mx);
            float alpha = __expf(m_run[v] - mnew);
            float rs = 0.f;
#pragma unroll
            for (int kg = 0; kg < 4; ++kg) {
                float p = __expf(sf[kg][v] - mnew);
                sf[kg][v] = p;
                rs += p;
            }
#pragma unroll
            for (int off = 1; off < 16; off <<= 1)
                rs += __shfl_xor(rs, off, 16);
            l_run[v] = l_run[v] * alpha + rs;
            m_run[v] = mnew;
#pragma unroll
            for (int ng = 0; ng < 4; ++ng) o[ng][v] *= alpha;
        }

        // restage P (C-frag -> A-frag) through per-wave LDS
#pragma unroll
        for (int kg = 0; kg < 4; ++kg)
#pragma unroll
            for (int v = 0; v < 8; ++v)
                Ps[wave][v + 8 * halfid][kg * 16 + lrow] =
                    (_Float16)sf[kg][v];

        // O += P(16x64) @ V(64x64)
#pragma unroll
        for (int c = 0; c < 2; ++c) {
            v16h pf = load_a_frag(&Ps[wave][0][0], 72, lrow, c * 32, kb);
#pragma unroll
            for (int ng = 0; ng < 4; ++ng) {
                v16h vf = load_b_frag(&Vs[buf][0][0], 72, ng * 16 + lrow,
                                      c * 32 + 16 * halfid);
                o[ng] = WMMA_F16(pf, vf, o[ng]);
            }
        }
    };

    stage(0, 0);
#pragma unroll 1
    for (int s0 = 0; s0 < S - 64; s0 += 64) {      // steady state: prefetch
        int buf = (s0 >> 6) & 1;
        stage(buf ^ 1, s0 + 64);
        async_wait_le<8>();        // current tile's 8 copies complete
        __syncthreads();
        compute_tile(buf);
        __syncthreads();
    }
    async_wait_le<0>();
    __syncthreads();
    compute_tile(((S >> 6) - 1) & 1);

    // normalize + write f16 ctx [B,S,D]
#pragma unroll
    for (int v = 0; v < 8; ++v) {
        float inv = 1.f / l_run[v];
        int s = q0 + v + 8 * halfid;
#pragma unroll
        for (int ng = 0; ng < 4; ++ng) {
            int d = ng * 16 + lrow;
            ctx[(size_t)(b * S + s) * D + h * Hd + d] =
                (_Float16)(o[ng][v] * inv);
        }
    }
}

extern "C" void kernel_launch(void* const* d_in, const int* in_sizes, int n_in,
                              void* d_out, int out_size, void* d_ws,
                              size_t ws_size, hipStream_t stream) {
    constexpr int B = CB, S = CS, D = CD;
    constexpr int M = CM;

    const float* x          = (const float*)d_in[0];
    const unsigned char* km = (const unsigned char*)d_in[1];
    const float* wq = (const float*)d_in[2];
    const float* bq = (const float*)d_in[3];
    const float* wk = (const float*)d_in[4];
    const float* bk = (const float*)d_in[5];
    const float* wv = (const float*)d_in[6];
    const float* bv = (const float*)d_in[7];
    const float* wo = (const float*)d_in[8];
    const float* bo = (const float*)d_in[9];
    float* out = (float*)d_out;

    // workspace (floats): xh[n/2] Qh[n/2] Kh[n/2] Vt[n/2] ctxh[n/2] whT[2M]
    size_t n = (size_t)M * D;
    float* W = (float*)d_ws;
    _Float16* xh   = (_Float16*)W;
    _Float16* Qh   = (_Float16*)(W + n / 2);
    _Float16* Kh   = (_Float16*)(W + n);
    _Float16* Vt   = (_Float16*)(W + 3 * n / 2);
    _Float16* ctxh = (_Float16*)(W + 2 * n);
    _Float16* wT   = (_Float16*)(W + 5 * n / 2);
    _Float16* wqT = wT;
    _Float16* wkT = wT + (size_t)D * D;
    _Float16* wvT = wT + 2 * (size_t)D * D;
    _Float16* woT = wT + 3 * (size_t)D * D;

    // one-time fp32 -> f16 conversions (weights transposed to [N][K])
    int n8 = (int)(n / 8);
    cvt_x_kernel<<<(n8 + 255) / 256, 256, 0, stream>>>(x, xh, n8);
    int wN = D * D;
    cvt_wT_kernel<<<(wN + 255) / 256, 256, 0, stream>>>(wq, wqT);
    cvt_wT_kernel<<<(wN + 255) / 256, 256, 0, stream>>>(wk, wkT);
    cvt_wT_kernel<<<(wN + 255) / 256, 256, 0, stream>>>(wv, wvT);
    cvt_wT_kernel<<<(wN + 255) / 256, 256, 0, stream>>>(wo, woT);

    dim3 gGemm(M / 128, D / 128);      // 64 x 8
    gemm_f16_async<0><<<gGemm, 256, 0, stream>>>(xh, wqT, bq, Qh, km);
    gemm_f16_async<0><<<gGemm, 256, 0, stream>>>(xh, wkT, bk, Kh, km);
    gemm_f16_async<1><<<gGemm, 256, 0, stream>>>(xh, wvT, bv, Vt, km);

    int ropeN = M * (D / 2);
    rope_inplace_kernel<<<(ropeN + 255) / 256, 256, 0, stream>>>(Qh, Kh);

    dim3 gAttn(S / 64, 16, B);         // 32 x 16 x 4
    attn_wmma<<<gAttn, 128, 0, stream>>>(Qh, Kh, Vt, km, ctxh);

    gemm_f16_async<2><<<gGemm, 256, 0, stream>>>(ctxh, woT, bo, out, km);
}
